// SwinBlock_21088289423773
// MI455X (gfx1250) — compile-verified, matrix operand order checked by reading, not run on hardware
//
#include <hip/hip_runtime.h>
#include <hip/hip_bf16.h>
#include <math.h>

typedef __bf16 bf16;
typedef __attribute__((ext_vector_type(16))) __bf16 v16bf;
typedef __attribute__((ext_vector_type(8)))  float  v8f;

#define Cc 96
#define Ww 224
#define HW (224*224)
#define NTOK (8*HW)   // 401408

// --- WMMA fragment helpers (wave32, v_wmma_f32_16x16x32_bf16) -------------
// A layout (ISA 7.12.2): lane&15 -> row, VGPR r holds K pairs:
//   kbase = (r>=4?16:0) + (lane>=16?8:0) + (r&3)*2
// B layout mirrors A with N in place of M.
__device__ __forceinline__ int kidx(int r, int lane) {
    return ((r & 4) << 2) | ((lane >> 4) << 3) | ((r & 3) << 1);
}

__device__ __forceinline__ v16bf load_frag(const bf16* base, int stride,
                                           int rowBase, int kBase, int lane) {
    v16bf f;
    const bf16* p = base + (rowBase + (lane & 15)) * stride + kBase;
#pragma unroll
    for (int r = 0; r < 8; ++r) {
        int kk = kidx(r, lane);
        f[2*r]   = p[kk];
        f[2*r+1] = p[kk+1];
    }
    return f;
}

__device__ __forceinline__ v8f wmma_bf16(v16bf a, v16bf b, v8f c) {
    return __builtin_amdgcn_wmma_f32_16x16x32_bf16(false, a, false, b,
                                                   (short)0, c, false, false);
}

// --- CDNA5 async global->LDS copy (ASYNCcnt-tracked, bypasses VGPRs) -------
__device__ __forceinline__ void async_b128(void* lds, const void* gsrc) {
    // generic LDS pointer: low 32 bits are the LDS offset (ISA 10.2 aperture)
    unsigned off = (unsigned)(unsigned long long)lds;
    asm volatile("global_load_async_to_lds_b128 %0, %1, off"
                 :: "v"(off), "v"(gsrc) : "memory");
}
__device__ __forceinline__ void async_copy(void* dst, const void* src,
                                           int n16, int tid) {
    char* d = (char*)dst; const char* s = (const char*)src;
    for (int e = tid; e < n16; e += 256)
        async_b128(d + e*16, s + e*16);
}
__device__ __forceinline__ void wait_async() {
    asm volatile("s_wait_asynccnt 0" ::: "memory");
}

// ---------------------------------------------------------------------------
// Kernel 0: weights -> bf16, transposed [N][K], padded K-stride so WMMA
// B-fragment K-pairs are contiguous. wq/wk/wv/wo->[96][100],
// w_fc1->[384][100], w_fc2->[96][392].
// ---------------------------------------------------------------------------
__global__ void __launch_bounds__(256) prep_weights(
        const float* __restrict__ wq, const float* __restrict__ wk,
        const float* __restrict__ wv, const float* __restrict__ wo,
        const float* __restrict__ wfc1, const float* __restrict__ wfc2,
        bf16* dwq, bf16* dwk, bf16* dwv, bf16* dwo, bf16* dfc1, bf16* dfc2) {
    int id = blockIdx.x * blockDim.x + threadIdx.x;
    if (id < 4*9600) {
        int a = id / 9600, rem = id % 9600;
        int n = rem / 100, kk = rem % 100;
        const float* src = (a==0)?wq:(a==1)?wk:(a==2)?wv:wo;
        bf16* dst        = (a==0)?dwq:(a==1)?dwk:(a==2)?dwv:dwo;
        dst[rem] = (bf16)((kk < 96) ? src[kk*96 + n] : 0.f);
    } else if (id < 4*9600 + 384*100) {
        int rem = id - 4*9600;
        int n = rem / 100, kk = rem % 100;
        dfc1[rem] = (bf16)((kk < 96) ? wfc1[kk*384 + n] : 0.f);
    } else if (id < 4*9600 + 38400 + 96*392) {
        int rem = id - 4*9600 - 38400;
        int n = rem / 392, kk = rem % 392;
        dfc2[rem] = (bf16)((kk < 384) ? wfc2[kk*96 + n] : 0.f);
    }
}

// ---------------------------------------------------------------------------
// Kernel 1: LN1 + transpose (B,C,H,W) -> token-major bf16 xn (NTOK,96).
// ---------------------------------------------------------------------------
__global__ void __launch_bounds__(256) ln1_kernel(
        const float* __restrict__ x, const float* __restrict__ g,
        const float* __restrict__ bta, bf16* __restrict__ xn) {
    __shared__ float xs[64][97];
    __shared__ float gs[96], bs[96];
    int tid = threadIdx.x;
    int nTiles = HW / 64;                       // 784
    int b = blockIdx.x / nTiles;
    int hwbase = (blockIdx.x % nTiles) * 64;
    const float* xb = x + (size_t)b * Cc * HW + hwbase;
    if (tid < 96) { gs[tid] = g[tid]; bs[tid] = bta[tid]; }
    for (int e = tid; e < 64*96; e += 256) {
        int c = e >> 6, p = e & 63;
        xs[p][c] = xb[(size_t)c * HW + p];
    }
    __syncthreads();
    if (tid < 64) {
        float m = 0.f;
        for (int c = 0; c < 96; ++c) m += xs[tid][c];
        m *= (1.f/96.f);
        float v = 0.f;
        for (int c = 0; c < 96; ++c) { float d = xs[tid][c]-m; v += d*d; }
        float inv = rsqrtf(v*(1.f/96.f) + 1e-5f);
        for (int c = 0; c < 96; ++c)
            xs[tid][c] = (xs[tid][c]-m)*inv*gs[c] + bs[c];
    }
    __syncthreads();
    unsigned int* dst = (unsigned int*)(xn + ((size_t)b*HW + hwbase) * 96);
    for (int u = tid; u < 64*48; u += 256) {
        int p = u / 48, cp = u % 48;
        unsigned short lo = __builtin_bit_cast(unsigned short, (bf16)xs[p][2*cp]);
        unsigned short hi = __builtin_bit_cast(unsigned short, (bf16)xs[p][2*cp+1]);
        dst[p*48 + cp] = ((unsigned int)hi << 16) | lo;
    }
}

// ---------------------------------------------------------------------------
// Kernel 2: windowed attention. One block per 7x7 window (8192 blocks).
// LDS layout (bytes):
//   [0,57600): wqT/wkT/wvT (phase1); phase2: scores f32 [64][66]@0,
//              attn bf16 [64][72]@16896
//   x/attnout bf16 [64][100]@57600, q bf16 [64][128]@70400, k@86784,
//   vT bf16 [4][32][72]@103168, gate@121600, redM@121856, redS@122880,
//   woT [96][100]@123904   -> total 143104
// ---------------------------------------------------------------------------
__global__ void __launch_bounds__(256) attn_kernel(
        const bf16* __restrict__ xn, const float* __restrict__ entropy,
        const float* __restrict__ beta,
        const bf16* __restrict__ wsWQ, const bf16* __restrict__ wsWK,
        const bf16* __restrict__ wsWV, const bf16* __restrict__ wsWO,
        const float* __restrict__ bq, const float* __restrict__ bk,
        const float* __restrict__ bv, const float* __restrict__ bo,
        float* __restrict__ attn_out) {
    extern __shared__ char smem[];
    bf16*  sWQ = (bf16*)(smem + 0);
    bf16*  sWK = (bf16*)(smem + 19200);
    bf16*  sWV = (bf16*)(smem + 38400);
    float* sSC = (float*)(smem + 0);
    bf16*  sAT = (bf16*)(smem + 16896);
    bf16*  sX  = (bf16*)(smem + 57600);
    bf16*  sAO = sX;                       // alias: x dead after QKV
    bf16*  sQ  = (bf16*)(smem + 70400);
    bf16*  sK  = (bf16*)(smem + 86784);
    bf16*  sVT = (bf16*)(smem + 103168);
    float* sG  = (float*)(smem + 121600);
    float* sRedM = (float*)(smem + 121856);
    float* sRedS = (float*)(smem + 122880);
    bf16*  sWO = (bf16*)(smem + 123904);

    int tid = threadIdx.x, wave = tid >> 5, lane = tid & 31;
    int hi8 = (lane >> 4) << 3;
    int b = blockIdx.x >> 10;
    int rem = blockIdx.x & 1023;
    int h0 = (rem >> 5) * 7, w0 = (rem & 31) * 7;

    // async-stage all four weight matrices (overlaps zero-fill + token load)
    async_copy(sWQ, wsWQ, 1200, tid);
    async_copy(sWK, wsWK, 1200, tid);
    async_copy(sWV, wsWV, 1200, tid);
    async_copy(sWO, wsWO, 1200, tid);

    { // zero x/q/k/vT span (row 49..63 pads, head-dim pads)
        unsigned int* z = (unsigned int*)(smem + 57600);
        for (int e = tid; e < 16000; e += 256) z[e] = 0u;
    }
    __syncthreads();
    { // load window tokens + gate
        const unsigned int* src = (const unsigned int*)xn;
        for (int e = tid; e < 49*48; e += 256) {
            int row = e / 48, d = e % 48;
            size_t tok = (size_t)b*HW + (size_t)(h0 + row/7)*Ww + (w0 + row%7);
            ((unsigned int*)sX)[row*50 + d] = src[tok*48 + d];
        }
        if (tid < 49) {
            float e = entropy[(size_t)b*HW + (size_t)(h0 + tid/7)*Ww + (w0 + tid%7)];
            sG[tid] = expf(-beta[0]*e);
        } else if (tid < 64) sG[tid] = 1.f;
    }
    wait_async();
    __syncthreads();

    // ---- QKV: three branch-free GEMM loops (64x96x96 each) ----
    for (int t = wave; t < 24; t += 8) {            // Q
        int m = t / 6, n = t % 6;
        v8f acc = {};
#pragma unroll
        for (int ks = 0; ks < 3; ++ks)
            acc = wmma_bf16(load_frag(sX, 100, m*16, ks*32, lane),
                            load_frag(sWQ, 100, n*16, ks*32, lane), acc);
        int col = n*16 + (lane & 15);
        int hd32 = col + (col/24)*8;                // head*32 + d
        float bval = bq[col];
#pragma unroll
        for (int r = 0; r < 8; ++r)
            sQ[(m*16 + r + hi8)*128 + hd32] = (bf16)(acc[r] + bval);
    }
    for (int t = wave; t < 24; t += 8) {            // K
        int m = t / 6, n = t % 6;
        v8f acc = {};
#pragma unroll
        for (int ks = 0; ks < 3; ++ks)
            acc = wmma_bf16(load_frag(sX, 100, m*16, ks*32, lane),
                            load_frag(sWK, 100, n*16, ks*32, lane), acc);
        int col = n*16 + (lane & 15);
        int hd32 = col + (col/24)*8;
        float bval = bk[col];
#pragma unroll
        for (int r = 0; r < 8; ++r)
            sK[(m*16 + r + hi8)*128 + hd32] = (bf16)(acc[r] + bval);
    }
    for (int t = wave; t < 24; t += 8) {            // V (stored key-major)
        int m = t / 6, n = t % 6;
        v8f acc = {};
#pragma unroll
        for (int ks = 0; ks < 3; ++ks)
            acc = wmma_bf16(load_frag(sX, 100, m*16, ks*32, lane),
                            load_frag(sWV, 100, n*16, ks*32, lane), acc);
        int col = n*16 + (lane & 15);
        int hd32 = col + (col/24)*8;
        float bval = bv[col];
#pragma unroll
        for (int r = 0; r < 8; ++r)
            sVT[hd32*72 + (m*16 + r + hi8)] = (bf16)(acc[r] + bval);
    }
    __syncthreads();

    const float scale = 0.20412414523193154f;   // 1/sqrt(24)
    for (int h = 0; h < 4; ++h) {
        // scores = q * k^T  (4x4 tiles, one 32-wide k-step per head)
        for (int t = wave; t < 16; t += 8) {
            int m = t >> 2, nt = t & 3;
            v8f acc = {};
            acc = wmma_bf16(load_frag(sQ, 128, m*16, h*32, lane),
                            load_frag(sK, 128, nt*16, h*32, lane), acc);
            int col = nt*16 + (lane & 15);
#pragma unroll
            for (int r = 0; r < 8; ++r)
                sSC[(m*16 + r + hi8)*66 + col] = acc[r] * scale;
        }
        __syncthreads();
        // softmax + gating, 4 lanes per row.
        // gating*renorm reduces to row scale g/(g+1e-9); pad rows get f=0.
        {
            int row = tid >> 2, part = tid & 3;
            int k0 = part * 13;
            int k1 = (part == 3) ? 49 : k0 + 13;
            float mx = -1e30f;
            for (int k2 = k0; k2 < k1; ++k2) mx = fmaxf(mx, sSC[row*66 + k2]);
            sRedM[(row<<2) | part] = mx;
            __syncthreads();
            float rmx = fmaxf(fmaxf(sRedM[row<<2], sRedM[(row<<2)|1]),
                              fmaxf(sRedM[(row<<2)|2], sRedM[(row<<2)|3]));
            float s = 0.f;
            for (int k2 = k0; k2 < k1; ++k2) s += expf(sSC[row*66 + k2] - rmx);
            sRedS[(row<<2) | part] = s;
            __syncthreads();
            float rs = sRedS[row<<2] + sRedS[(row<<2)|1]
                     + sRedS[(row<<2)|2] + sRedS[(row<<2)|3];
            float g = sG[row];
            float f = (row < 49) ? g / ((g + 1e-9f) * rs) : 0.f;
            for (int k2 = k0; k2 < k1; ++k2)
                sAT[row*72 + k2] = (bf16)(expf(sSC[row*66 + k2] - rmx) * f);
            int z0 = 49 + part*4, z1 = (part == 3) ? 64 : z0 + 4;
            for (int k2 = z0; k2 < z1; ++k2) sAT[row*72 + k2] = (bf16)0.f;
        }
        __syncthreads();
        // out = attn * v  (4x2 tiles == 8 waves, K=64 keys -> 2 steps)
        {
            int m = wave >> 1, nt = wave & 1;
            v8f acc = {};
#pragma unroll
            for (int ks = 0; ks < 2; ++ks)
                acc = wmma_bf16(load_frag(sAT, 72, m*16, ks*32, lane),
                                load_frag(sVT + h*32*72, 72, nt*16, ks*32, lane), acc);
            int d = nt*16 + (lane & 15);
            if (d < 24) {
#pragma unroll
                for (int r = 0; r < 8; ++r)
                    sAO[(m*16 + r + hi8)*100 + h*24 + d] = (bf16)acc[r];
            }
        }
        __syncthreads();
    }

    // ---- output projection + store (residual fused into MLP kernel) ----
    for (int t = wave; t < 24; t += 8) {
        int m = t / 6, n = t % 6;
        v8f acc = {};
#pragma unroll
        for (int ks = 0; ks < 3; ++ks)
            acc = wmma_bf16(load_frag(sAO, 100, m*16, ks*32, lane),
                            load_frag(sWO, 100, n*16, ks*32, lane), acc);
        int col = n*16 + (lane & 15);
        float bval = bo[col];
#pragma unroll
        for (int r = 0; r < 8; ++r) {
            int row = m*16 + r + hi8;
            if (row < 49) {
                size_t tok = (size_t)b*HW + (size_t)(h0 + row/7)*Ww + (w0 + row%7);
                attn_out[tok*96 + col] = acc[r] + bval;
            }
        }
    }
}

// ---------------------------------------------------------------------------
// Kernel 3: residual + LN2 + fc1 + GELU + fc2 + residual + (B,C,H,W) store.
// LDS: fc1T/f32-staging [0,76800), xn2 [64][100]@76800, h [64][392]@89600,
// xres f32 [64][97]@139776, fc2T [96][392]@164608 -> total 239872.
// ---------------------------------------------------------------------------
__global__ void __launch_bounds__(256) mlp_kernel(
        const float* __restrict__ x, const float* __restrict__ attn_out,
        const float* __restrict__ g2, const float* __restrict__ b2,
        const bf16* __restrict__ wsFC1, const float* __restrict__ bfc1,
        const bf16* __restrict__ wsFC2, const float* __restrict__ bfc2,
        float* __restrict__ out) {
    extern __shared__ char smem[];
    bf16*  sW1 = (bf16*)(smem + 0);
    float* sStage = (float*)(smem + 0);
    bf16*  sXN = (bf16*)(smem + 76800);
    bf16*  sH  = (bf16*)(smem + 89600);
    float* sR  = (float*)(smem + 139776);
    bf16*  sW2 = (bf16*)(smem + 164608);

    int tid = threadIdx.x, wave = tid >> 5, lane = tid & 31;
    int hi8 = (lane >> 4) << 3;
    int nTiles = HW / 64;
    int b = blockIdx.x / nTiles;
    int hwbase = (blockIdx.x % nTiles) * 64;
    const float* xb = x + (size_t)b*Cc*HW + hwbase;
    const float* ao = attn_out + ((size_t)b*HW + hwbase) * 96;

    async_copy(sW1, wsFC1, 4800, tid);               // 76800 B
    async_copy(sW2, wsFC2, 4704, tid);               // 75264 B

    for (int e = tid; e < 64*96; e += 256) {          // x (channel-major)
        int c = e >> 6, p = e & 63;
        sR[p*97 + c] = xb[(size_t)c*HW + p];
    }
    __syncthreads();
    for (int e = tid; e < 64*96; e += 256) {          // + attn_out (row-major)
        int p = e / 96, c = e % 96;
        sR[p*97 + c] += ao[(size_t)p*96 + c];
    }
    wait_async();
    __syncthreads();
    if (tid < 64) {                                   // LN2
        float m = 0.f;
        for (int c = 0; c < 96; ++c) m += sR[tid*97 + c];
        m *= (1.f/96.f);
        float v = 0.f;
        for (int c = 0; c < 96; ++c) { float d = sR[tid*97+c]-m; v += d*d; }
        float inv = rsqrtf(v*(1.f/96.f) + 1e-5f);
        for (int c = 0; c < 96; ++c)
            sXN[tid*100 + c] = (bf16)((sR[tid*97+c]-m)*inv*g2[c] + b2[c]);
    }
    __syncthreads();
    // fc1 + exact GELU: 4x24 tiles over 8 waves
    for (int t = wave; t < 96; t += 8) {
        int m = t / 24, n = t % 24;
        v8f acc = {};
#pragma unroll
        for (int ks = 0; ks < 3; ++ks)
            acc = wmma_bf16(load_frag(sXN, 100, m*16, ks*32, lane),
                            load_frag(sW1, 100, n*16, ks*32, lane), acc);
        int col = n*16 + (lane & 15);
        float bias = bfc1[col];
#pragma unroll
        for (int r = 0; r < 8; ++r) {
            float v = acc[r] + bias;
            v = 0.5f * v * (1.f + erff(v * 0.70710678118654752f));
            sH[(m*16 + r + hi8)*392 + col] = (bf16)v;
        }
    }
    __syncthreads();
    // fc2 + bias + residual; hold 3 tiles/wave in registers across barrier
    v8f fin[3]; int cols[3], mrows[3];
#pragma unroll
    for (int ti = 0; ti < 3; ++ti) {
        int t = wave + ti*8;
        int m = t / 6, n = t % 6;
        v8f acc = {};
#pragma unroll
        for (int ks = 0; ks < 12; ++ks)
            acc = wmma_bf16(load_frag(sH, 392, m*16, ks*32, lane),
                            load_frag(sW2, 392, n*16, ks*32, lane), acc);
        int col = n*16 + (lane & 15);
        float bias = bfc2[col];
#pragma unroll
        for (int r = 0; r < 8; ++r)
            acc[r] = acc[r] + bias + sR[(m*16 + r + hi8)*97 + col];
        fin[ti] = acc; cols[ti] = col; mrows[ti] = m*16;
    }
    __syncthreads();
#pragma unroll
    for (int ti = 0; ti < 3; ++ti)
#pragma unroll
        for (int r = 0; r < 8; ++r)
            sStage[(mrows[ti] + r + hi8)*97 + cols[ti]] = fin[ti][r];
    __syncthreads();
    float* ob = out + (size_t)b*Cc*HW + hwbase;       // (B,C,H,W) coalesced
    for (int e = tid; e < 64*96; e += 256) {
        int c = e >> 6, p = e & 63;
        ob[(size_t)c*HW + p] = sStage[p*97 + c];
    }
}

// ---------------------------------------------------------------------------
extern "C" void kernel_launch(void* const* d_in, const int* in_sizes, int n_in,
                              void* d_out, int out_size, void* d_ws, size_t ws_size,
                              hipStream_t stream) {
    (void)in_sizes; (void)n_in; (void)out_size; (void)ws_size;
    const float* x   = (const float*)d_in[0];
    const float* ent = (const float*)d_in[1];
    const float* bet = (const float*)d_in[2];
    const float* wq  = (const float*)d_in[3];
    const float* bq  = (const float*)d_in[4];
    const float* wk  = (const float*)d_in[5];
    const float* bk  = (const float*)d_in[6];
    const float* wv  = (const float*)d_in[7];
    const float* bv  = (const float*)d_in[8];
    const float* wo  = (const float*)d_in[9];
    const float* bo  = (const float*)d_in[10];
    const float* g1  = (const float*)d_in[11];
    const float* b1  = (const float*)d_in[12];
    const float* g2  = (const float*)d_in[13];
    const float* b2  = (const float*)d_in[14];
    const float* wf1 = (const float*)d_in[15];
    const float* bf1 = (const float*)d_in[16];
    const float* wf2 = (const float*)d_in[17];
    const float* bf2 = (const float*)d_in[18];
    float* out = (float*)d_out;

    char* ws = (char*)d_ws;
    const size_t OFF_XN  = 0;                           // bf16 NTOK*96
    const size_t OFF_AO  = (size_t)NTOK*96*2;           // f32  NTOK*96
    const size_t OFF_WQ  = OFF_AO + (size_t)NTOK*96*4;  // 16B aligned
    const size_t OFF_WK  = OFF_WQ + 19200;
    const size_t OFF_WV  = OFF_WK + 19200;
    const size_t OFF_WO  = OFF_WV + 19200;
    const size_t OFF_F1  = OFF_WO + 19200;
    const size_t OFF_F2  = OFF_F1 + 76800;

    bf16*  dXN = (bf16*)(ws + OFF_XN);
    float* dAO = (float*)(ws + OFF_AO);
    bf16*  dWQ = (bf16*)(ws + OFF_WQ);
    bf16*  dWK = (bf16*)(ws + OFF_WK);
    bf16*  dWV = (bf16*)(ws + OFF_WV);
    bf16*  dWO = (bf16*)(ws + OFF_WO);
    bf16*  dF1 = (bf16*)(ws + OFF_F1);
    bf16*  dF2 = (bf16*)(ws + OFF_F2);

    prep_weights<<<(114432 + 255)/256, 256, 0, stream>>>(
        wq, wk, wv, wo, wf1, wf2, dWQ, dWK, dWV, dWO, dF1, dF2);
    ln1_kernel<<<NTOK/64, 256, 0, stream>>>(x, g1, b1, dXN);
    attn_kernel<<<8192, 256, 143104, stream>>>(
        dXN, ent, bet, dWQ, dWK, dWV, dWO, bq, bk, bv, bo, dAO);
    mlp_kernel<<<NTOK/64, 256, 239872, stream>>>(
        x, dAO, g2, b2, dF1, bf1, dF2, bf2, out);
}
